// Pgwnet_66331474919442
// MI455X (gfx1250) — matrix operand order repaired and checked by reading, not compiled, not run on hardware
//
#include <hip/hip_runtime.h>

typedef __attribute__((ext_vector_type(16))) _Float16 v16h;
typedef __attribute__((ext_vector_type(8)))  _Float16 v8h;
typedef __attribute__((ext_vector_type(8)))  float    v8f;

#define NB   325
#define BB   64
#define T0   12
#define REC  13
#define RCC  32
#define SCC  256
#define ECC  512
#define OUTC 12
#define GINC 224
#define NPOI 14
#define KBIG (1<<30)

union HF { v16h v; v8h h8[2]; _Float16 h[16]; };
union AF { v8f  v; float f[8]; };
union H2 { _Float16 h[2]; unsigned int u; };

// ---------------------------------------------------------------------------
// Generic WMMA GEMM:  C[m,ncol] (op) = act( sum_k A(m,k) * W(ncol,k) + bias )
//   A(m,k) = Ab[ (m + (m/P)*Q + R + (k/Ktap)*Dtap) * lda + (k % Ktap) ]
//   W(ncol,k) = Wb[ k*wsK + ncol*wsN ]
//   op: 0=store, 1=accumulate into C, 2=relu-store
// Block: 128 threads (4 waves), C-tile 128(M) x 32(N); wave = 32x32 (4 wmma).
// Double-buffered LDS, one barrier per K-step, hoisted addresses,
// packed b32 LDS staging (two f16 per store).
// ---------------------------------------------------------------------------
__global__ __launch_bounds__(128)
void wmma_gemm(const float* __restrict__ A, long aBatch, int lda,
               int P, int Q, int R, int Ktap, int Dtap,
               const float* __restrict__ W, long wBatch, long wsK, long wsN,
               const float* __restrict__ bias,
               float* __restrict__ C, long cBatch, int ldc,
               int M, int Nc, int K, int op)
{
    __shared__ _Float16 As[2][128][40];   // rows = m-local, cols = k-local (80B stride)
    __shared__ _Float16 Bs[2][32][40];    // rows = n-local, cols = k-local

    const int tid    = threadIdx.x;
    const int wave   = tid >> 5;
    const int lane   = tid & 31;
    const int laneLo = lane & 15;
    const int hi     = lane >> 4;
    const int dcol   = (tid & 15) * 2;    // even k-column pair owned by this thread
    const int rbase  = tid >> 4;          // 0..7 ; staged rows step by 8 per it
    const int z      = blockIdx.z;
    const int mBlock = blockIdx.x * 128;
    const int nBlock = blockIdx.y * 32;

    const float* Ab = A + (long)z * aBatch;
    const float* Wb = W + (long)z * wBatch;
    float*       Cb = C + (long)z * cBatch;

    const bool bigK = (Ktap >= K);        // linear-K mode (Ktap never splits)

    // ---- hoisted staging addresses (element offsets, all < 2^31) ----
    int  aBase[16]; bool aMv[16];
    #pragma unroll
    for (int it = 0; it < 16; ++it) {
        int m  = mBlock + it * 8 + rbase;
        bool v = (m < M);
        int mm = v ? m : 0;
        long sr = (long)mm + (long)(mm / P) * Q + R;
        aBase[it] = (int)(sr * lda + dcol);
        aMv[it]   = v;
    }
    int  bBase[4]; bool bMv[4];
    #pragma unroll
    for (int it = 0; it < 4; ++it) {
        int ncol = nBlock + it * 8 + rbase;
        bool v = (ncol < Nc);
        int nn = v ? ncol : 0;
        bBase[it] = (int)((long)dcol * wsK + (long)nn * wsN);
        bMv[it]   = v;
    }

    float ra0[16], ra1[16], rb0[4], rb1[4];
    auto loadTiles = [&](int k0) {
        bool k0Ok = (k0 + dcol)     < K;
        bool k1Ok = (k0 + dcol + 1) < K;
        int aAdd = bigK ? k0 : (k0 / Ktap) * Dtap * lda;
        int bAdd = (int)((long)k0 * wsK);
        int wK   = (int)wsK;
        #pragma unroll
        for (int it = 0; it < 16; ++it) {
            bool v0 = aMv[it] && k0Ok;
            bool v1 = aMv[it] && k1Ok;
            int a0 = v0 ? (aBase[it] + aAdd)     : 0;
            int a1 = v1 ? (aBase[it] + aAdd + 1) : 0;
            float x0 = Ab[a0], x1 = Ab[a1];
            ra0[it] = v0 ? x0 : 0.f;
            ra1[it] = v1 ? x1 : 0.f;
        }
        #pragma unroll
        for (int it = 0; it < 4; ++it) {
            bool v0 = bMv[it] && k0Ok;
            bool v1 = bMv[it] && k1Ok;
            int a0 = v0 ? (bBase[it] + bAdd)      : 0;
            int a1 = v1 ? (bBase[it] + bAdd + wK) : 0;
            float x0 = Wb[a0], x1 = Wb[a1];
            rb0[it] = v0 ? x0 : 0.f;
            rb1[it] = v1 ? x1 : 0.f;
        }
    };
    auto storeTiles = [&](int p) {
        #pragma unroll
        for (int it = 0; it < 16; ++it) {
            H2 t; t.h[0] = (_Float16)ra0[it]; t.h[1] = (_Float16)ra1[it];
            *(unsigned int*)&As[p][it * 8 + rbase][dcol] = t.u;
        }
        #pragma unroll
        for (int it = 0; it < 4; ++it) {
            H2 t; t.h[0] = (_Float16)rb0[it]; t.h[1] = (_Float16)rb1[it];
            *(unsigned int*)&Bs[p][it * 8 + rbase][dcol] = t.u;
        }
    };

    AF acc[2][2];
    acc[0][0].v = (v8f){0.f,0.f,0.f,0.f,0.f,0.f,0.f,0.f};
    acc[0][1].v = acc[0][0].v; acc[1][0].v = acc[0][0].v; acc[1][1].v = acc[0][0].v;

    loadTiles(0);
    storeTiles(0);
    __syncthreads();
    int p = 0;
    for (int k0 = 0; k0 < K; k0 += 32) {
        const bool more = (k0 + 32) < K;
        if (more) loadTiles(k0 + 32);

        // B fragments (contiguous k order, same as A): lane n = laneLo, K = 16*hi + e
        HF b0, b1;
        {
            const _Float16* br0 = &Bs[p][laneLo][0];
            const _Float16* br1 = &Bs[p][16 + laneLo][0];
            b0.h8[0] = *(const v8h*)(br0 + 16 * hi);
            b0.h8[1] = *(const v8h*)(br0 + 16 * hi + 8);
            b1.h8[0] = *(const v8h*)(br1 + 16 * hi);
            b1.h8[1] = *(const v8h*)(br1 + 16 * hi + 8);
        }
        // A fragments: lane m = laneLo, K = 16*hi + e (matches B)
        #pragma unroll
        for (int w = 0; w < 2; ++w) {
            HF af;
            const _Float16* ar = &As[p][wave * 32 + w * 16 + laneLo][0];
            af.h8[0] = *(const v8h*)(ar + 16 * hi);
            af.h8[1] = *(const v8h*)(ar + 16 * hi + 8);
            acc[w][0].v = __builtin_amdgcn_wmma_f32_16x16x32_f16(false, af.v, false, b0.v,
                                                                 (short)0, acc[w][0].v, false, false);
            acc[w][1].v = __builtin_amdgcn_wmma_f32_16x16x32_f16(false, af.v, false, b1.v,
                                                                 (short)0, acc[w][1].v, false, false);
        }
        if (more) {
            storeTiles(p ^ 1);
            __syncthreads();
            p ^= 1;
        }
    }

    // store: C/D layout -> M = r + 8*hi, N = laneLo (+16 for tile 1)
    #pragma unroll
    for (int w = 0; w < 2; ++w) {
        #pragma unroll
        for (int r = 0; r < 8; ++r) {
            int mrow = mBlock + wave * 32 + w * 16 + r + 8 * hi;
            if (mrow >= M) continue;
            long base = (long)mrow * ldc;
            int nc0 = nBlock + laneLo;
            int nc1 = nBlock + 16 + laneLo;
            float v0 = acc[w][0].f[r], v1 = acc[w][1].f[r];
            if (bias) {
                if (nc0 < Nc) v0 += bias[nc0];
                if (nc1 < Nc) v1 += bias[nc1];
            }
            if (nc0 < Nc) {
                float o = v0;
                if (op == 1) o += Cb[base + nc0];
                if (op == 2) o = fmaxf(o, 0.f);
                Cb[base + nc0] = o;
            }
            if (nc1 < Nc) {
                float o = v1;
                if (op == 1) o += Cb[base + nc1];
                if (op == 2) o = fmaxf(o, 0.f);
                Cb[base + nc1] = o;
            }
        }
    }
}

// ---------------------------------------------------------------------------
// Elementwise / reduction helpers
// ---------------------------------------------------------------------------
__global__ void k_zero(float* p, int n) {
    int i = blockIdx.x * 256 + threadIdx.x;
    if (i < n) p[i] = 0.f;
}

__global__ void k_redstats(const float* __restrict__ x, int n, float* stats) {
    __shared__ float s0[256], s1[256];
    int tid = threadIdx.x;
    float a = 0.f, b = 0.f;
    for (long i = blockIdx.x * 256 + tid; i < n; i += (long)gridDim.x * 256) {
        float v = x[i]; a += v; b += v * v;
    }
    s0[tid] = a; s1[tid] = b; __syncthreads();
    for (int s = 128; s > 0; s >>= 1) {
        if (tid < s) { s0[tid] += s0[tid + s]; s1[tid] += s1[tid + s]; }
        __syncthreads();
    }
    if (tid == 0) { atomicAdd(&stats[0], s0[0]); atomicAdd(&stats[1], s1[0]); }
}

// input BN + front pad to REC + start 1x1 conv; x0 layout [B,REC,N,32]
__global__ void k_start(const float* __restrict__ in, const float* __restrict__ sW,
                        const float* __restrict__ sb, const float* __restrict__ stats,
                        float cnt, float* __restrict__ x0) {
    int i = blockIdx.x * 256 + threadIdx.x;
    int total = BB * REC * NB * RCC;
    if (i >= total) return;
    int o = i & 31; int r = i >> 5;
    int n = r % NB; int bt = r / NB; int t = bt % REC; int b = bt / REC;
    float v = 0.f;
    if (t >= 1) {
        float m   = stats[0] / cnt;
        float var = stats[1] / cnt - m * m;
        float xin = in[((long)b * NB + n) * T0 + (t - 1)];
        v = (xin - m) * rsqrtf(var + 1e-5f);
    }
    x0[i] = sW[o] * v + sb[o];
}

// AT[s][w][v] = sup[s][v][w] for s=0,1
__global__ void k_tsup(const float* __restrict__ sup, float* __restrict__ AT) {
    int i = blockIdx.x * 256 + threadIdx.x;
    int total = 2 * NB * NB;
    if (i >= total) return;
    int s = i / (NB * NB); int rem = i % (NB * NB);
    int v = rem / NB; int w = rem % NB;
    AT[(long)s * NB * NB + (long)w * NB + v] = sup[i];
}

// adp = softmax(relu(nv1 @ nv2), axis=1), written transposed into AT slot 2
__global__ __launch_bounds__(128)
void k_adp(const float* __restrict__ nv1, const float* __restrict__ nv2,
           float* __restrict__ AT2) {
    __shared__ float sl[NB];
    __shared__ float red[128];
    int r = blockIdx.x, tid = threadIdx.x;
    float n1l[10];
    #pragma unroll
    for (int d = 0; d < 10; ++d) n1l[d] = nv1[r * 10 + d];
    float lmax = -1e30f;
    for (int j = tid; j < NB; j += 128) {
        float a = 0.f;
        #pragma unroll
        for (int d = 0; d < 10; ++d) a += n1l[d] * nv2[d * NB + j];
        a = fmaxf(a, 0.f);
        sl[j] = a; lmax = fmaxf(lmax, a);
    }
    red[tid] = lmax; __syncthreads();
    for (int s = 64; s > 0; s >>= 1) { if (tid < s) red[tid] = fmaxf(red[tid], red[tid + s]); __syncthreads(); }
    float mx = red[0]; __syncthreads();
    float lsum = 0.f;
    for (int j = tid; j < NB; j += 128) { float e = expf(sl[j] - mx); sl[j] = e; lsum += e; }
    red[tid] = lsum; __syncthreads();
    for (int s = 64; s > 0; s >>= 1) { if (tid < s) red[tid] += red[tid + s]; __syncthreads(); }
    float inv = 1.f / red[0];
    for (int j = tid; j < NB; j += 128) AT2[(long)j * NB + r] = sl[j] * inv;
}

// repack (L,32,32,1,2) -> [L][o][tap*32+cin]
__global__ void k_repack(const float* __restrict__ w, float* __restrict__ o) {
    int i = blockIdx.x * 256 + threadIdx.x;
    if (i >= 8 * 2048) return;
    int l = i / 2048; int r = i % 2048;
    int oc = r / 64; int kk = r % 64;
    int tap = kk >> 5; int c = kk & 31;
    o[i] = w[(((long)l * 32 + oc) * 32 + c) * 2 + tap];
}

__global__ void k_gate(const float* __restrict__ f, const float* __restrict__ g,
                       float* __restrict__ xg, int n) {
    int i = blockIdx.x * 256 + threadIdx.x;
    if (i >= n) return;
    xg[i] = tanhf(f[i]) * (1.f / (1.f + expf(-g[i])));
}

// xn += residual (row-remapped), accumulate per-channel sum/sumsq (c = tid&31)
__global__ void k_resid_stats(float* __restrict__ xn, const float* __restrict__ xp,
                              int P, int Q, int R, int n, float* stats) {
    __shared__ float s0[256], s1[256];
    int tid = threadIdx.x;
    float a = 0.f, b = 0.f;
    for (long i = blockIdx.x * 256 + tid; i < n; i += (long)gridDim.x * 256) {
        int c = (int)(i & 31); long m = i >> 5;
        long src = (m + (m / P) * (long)Q + R) * RCC + c;
        float v = xn[i] + xp[src];
        xn[i] = v; a += v; b += v * v;
    }
    s0[tid] = a; s1[tid] = b; __syncthreads();
    for (int s = 128; s >= 32; s >>= 1) {
        if (tid < s) { s0[tid] += s0[tid + s]; s1[tid] += s1[tid + s]; }
        __syncthreads();
    }
    if (tid < 32) { atomicAdd(&stats[2 + tid], s0[tid]); atomicAdd(&stats[34 + tid], s1[tid]); }
}

__global__ void k_bn_apply(const float* __restrict__ xn, const float* __restrict__ stats,
                           const float* __restrict__ gam, const float* __restrict__ bet,
                           float cnt, float* __restrict__ xo, int n) {
    int i = blockIdx.x * 256 + threadIdx.x;
    if (i >= n) return;
    int c = i & 31;
    float m   = stats[2 + c] / cnt;
    float var = stats[34 + c] / cnt - m * m;
    xo[i] = (xn[i] - m) * rsqrtf(var + 1e-5f) * gam[c] + bet[c];
}

__global__ void k_relu(const float* __restrict__ x, float* __restrict__ y, int n) {
    int i = blockIdx.x * 256 + threadIdx.x;
    if (i < n) y[i] = fmaxf(x[i], 0.f);
}

// km[p][d] = sum_j poiVec[p][j] * Wk[j][d]
__global__ void k_kmat(const float* __restrict__ pv, const float* __restrict__ Wk,
                       float* __restrict__ km) {
    int i = blockIdx.x * 256 + threadIdx.x;
    if (i >= NPOI * SCC) return;
    int p = i / SCC, d = i % SCC;
    float a = 0.f;
    #pragma unroll
    for (int j = 0; j < 32; ++j) a += pv[p * 32 + j] * Wk[j * SCC + d];
    km[i] = a;
}

__global__ void k_attn(const float* __restrict__ q, const float* __restrict__ km,
                       const float* __restrict__ poi, float* __restrict__ dist,
                       float* __restrict__ po_out, float* __restrict__ pobuf) {
    int i = blockIdx.x * 256 + threadIdx.x;
    if (i >= BB * NB) return;
    int n = i % NB;
    const float* qr = q + (long)i * SCC;
    float sc[NPOI]; float mx = -1e30f;
    for (int p = 0; p < NPOI; ++p) {
        float a = 0.f;
        for (int d = 0; d < SCC; ++d) a += qr[d] * km[p * SCC + d];
        a *= (1.f / 16.f);      // 1/sqrt(256)
        sc[p] = a; mx = fmaxf(mx, a);
    }
    float sum = 0.f;
    for (int p = 0; p < NPOI; ++p) { sc[p] = expf(sc[p] - mx); sum += sc[p]; }
    float inv = 1.f / sum, acc = 0.f;
    for (int p = 0; p < NPOI; ++p) {
        float d = sc[p] * inv;
        dist[(long)i * NPOI + p] = d;
        acc += d * poi[n * NPOI + p];
    }
    float po = 1.f / (1.f + expf(-acc));
    po_out[i] = po; pobuf[i] = po;
}

__global__ void k_scale(const float* __restrict__ xr, const float* __restrict__ po,
                        float* __restrict__ xs, int n) {
    int i = blockIdx.x * 256 + threadIdx.x;
    if (i >= n) return;
    xs[i] = xr[i] * (1.f + po[i >> 8]);   // 256 channels
}

// out[b][c][n] = e2[(b*N+n)*12 + c]
__global__ void k_outT(const float* __restrict__ e2, float* __restrict__ out, int n) {
    int i = blockIdx.x * 256 + threadIdx.x;
    if (i >= n) return;
    int b = i / (OUTC * NB); int r = i % (OUTC * NB);
    int c = r / NB; int nn = r % NB;
    out[i] = e2[((long)b * NB + nn) * OUTC + c];
}

// ---------------------------------------------------------------------------
static inline void gemm(hipStream_t st, const float* A, long aB, int lda,
                        int P, int Q, int R, int Ktap, int Dtap,
                        const float* W, long wB, long wsK, long wsN, const float* bias,
                        float* C, long cB, int ldc, int M, int Nc, int K, int op, int bat) {
    dim3 g((M + 127) / 128, (Nc + 31) / 32, bat);
    wmma_gemm<<<g, 128, 0, st>>>(A, aB, lda, P, Q, R, Ktap, Dtap,
                                 W, wB, wsK, wsN, bias, C, cB, ldc, M, Nc, K, op);
}
static inline dim3 ew(long n) { long b = (n + 255) / 256; if (b > 1048576) b = 1048576; return dim3((unsigned)b); }
static inline dim3 gs(long n) { long b = (n + 255) / 256; if (b > 2048) b = 2048; return dim3((unsigned)b); }

extern "C" void kernel_launch(void* const* d_in, const int* in_sizes, int n_in,
                              void* d_out, int out_size, void* d_ws, size_t ws_size,
                              hipStream_t stream) {
    const float* in     = (const float*)d_in[0];
    const float* sup    = (const float*)d_in[1];
    const float* nv1    = (const float*)d_in[2];
    const float* nv2    = (const float*)d_in[3];
    const float* startW = (const float*)d_in[4];
    const float* startB = (const float*)d_in[5];
    const float* filtW  = (const float*)d_in[6];
    const float* filtB  = (const float*)d_in[7];
    const float* gateW  = (const float*)d_in[8];
    const float* gateB  = (const float*)d_in[9];
    const float* skipW  = (const float*)d_in[10];
    const float* skipB  = (const float*)d_in[11];
    const float* gcW    = (const float*)d_in[12];
    const float* gcB    = (const float*)d_in[13];
    const float* bnG    = (const float*)d_in[14];
    const float* bnB    = (const float*)d_in[15];
    const float* e1W    = (const float*)d_in[16];
    const float* e1B    = (const float*)d_in[17];
    const float* e2W    = (const float*)d_in[18];
    const float* e2B    = (const float*)d_in[19];
    const float* Wq     = (const float*)d_in[20];
    const float* Wk     = (const float*)d_in[21];
    const float* poi    = (const float*)d_in[22];
    const float* poiVec = (const float*)d_in[23];
    float* out = (float*)d_out;
    float* ws  = (float*)d_ws;

    // workspace layout (floats)
    const size_t XA_SZ = (size_t)BB * REC * NB * RCC;   // 8,652,800
    const size_t XG_SZ = (size_t)BB * T0 * NB * RCC;    // 7,987,200
    const size_t SK_SZ = (size_t)BB * NB * SCC;         // 5,324,800
    const size_t NN    = (size_t)NB * NB;
    size_t off = 0;
    float* xA    = ws + off; off += XA_SZ;
    float* xB    = ws + off; off += XA_SZ;
    float* xg    = ws + off; off += XG_SZ;
    float* t1    = ws + off; off += XG_SZ;
    float* t2    = ws + off; off += XG_SZ;
    float* xn    = ws + off; off += XG_SZ;
    float* skip  = ws + off; off += SK_SZ;
    float* AT    = ws + off; off += 3 * NN;
    float* stats = ws + off; off += 128;
    float* fwRe  = ws + off; off += 8 * 2048;
    float* gwRe  = ws + off; off += 8 * 2048;
    float* kmat  = ws + off; off += NPOI * SCC;
    float* pobuf = ws + off; off += (size_t)BB * NB;
    float* e2buf = ws + off; off += (size_t)BB * NB * OUTC;
    if (ws_size < off * sizeof(float)) return;   // deterministic early-out
    float* e1buf = xA;                           // layers done by then; xA+xB contiguous

    // --- prep ---
    k_zero<<<ew(128), 256, 0, stream>>>(stats, 128);
    k_redstats<<<gs(BB * NB * T0), 256, 0, stream>>>(in, BB * NB * T0, stats);
    k_start<<<ew((long)XA_SZ), 256, 0, stream>>>(in, startW, startB, stats,
                                                 (float)(BB * NB * T0), xA);
    k_tsup<<<ew(2 * NN), 256, 0, stream>>>(sup, AT);
    k_adp<<<NB, 128, 0, stream>>>(nv1, nv2, AT + 2 * NN);
    k_repack<<<ew(8 * 2048), 256, 0, stream>>>(filtW, fwRe);
    k_repack<<<ew(8 * 2048), 256, 0, stream>>>(gateW, gwRe);

    // --- layers ---
    int Tin = REC;
    float* xcur = xA;
    float* xnxt = xB;
    for (int l = 0; l < 8; ++l) {
        int dil  = (l & 1) ? 2 : 1;
        int Tout = Tin - dil;
        int Mc   = BB * Tout * NB;

        // dilated k=2 convs (two taps folded into K=64)
        gemm(stream, xcur, 0, RCC, Tout * NB, (Tin - Tout) * NB, 0, 32, dil * NB,
             fwRe + l * 2048, 0, 1, 64, filtB + l * 32, t1, 0, RCC, Mc, RCC, 64, 0, 1);
        gemm(stream, xcur, 0, RCC, Tout * NB, (Tin - Tout) * NB, 0, 32, dil * NB,
             gwRe + l * 2048, 0, 1, 64, gateB + l * 32, t2, 0, RCC, Mc, RCC, 64, 0, 1);
        k_gate<<<ew((long)Mc * RCC), 256, 0, stream>>>(t1, t2, xg, Mc * RCC);

        // skip: only last time slice survives to the head
        gemm(stream, xg, 0, RCC, NB, (Tout - 1) * NB, (Tout - 1) * NB, KBIG, 0,
             skipW + (long)l * SCC * RCC, 0, 1, RCC, skipB + l * SCC,
             skip, 0, SCC, BB * NB, SCC, RCC, (l == 0 ? 0 : 1), 1);

        // gconv: xn = b + W[:,0:32] x  (+ per-support hop chunks)
        gemm(stream, xg, 0, RCC, KBIG, 0, 0, KBIG, 0,
             gcW + (long)l * RCC * GINC, 0, 1, GINC, gcB + l * 32,
             xn, 0, RCC, Mc, RCC, RCC, 0, 1);
        for (int s = 0; s < 3; ++s) {
            // t1 = A_s^T @ x   (batched over B*Tout)
            gemm(stream, AT + (long)s * NN, 0, NB, KBIG, 0, 0, KBIG, 0,
                 xg, (long)NB * RCC, RCC, 1, nullptr,
                 t1, (long)NB * RCC, RCC, NB, RCC, NB, 0, BB * Tout);
            gemm(stream, t1, 0, RCC, KBIG, 0, 0, KBIG, 0,
                 gcW + (long)l * RCC * GINC + (32 + 64 * s), 0, 1, GINC, nullptr,
                 xn, 0, RCC, Mc, RCC, RCC, 1, 1);
            // t2 = A_s^T @ t1
            gemm(stream, AT + (long)s * NN, 0, NB, KBIG, 0, 0, KBIG, 0,
                 t1, (long)NB * RCC, RCC, 1, nullptr,
                 t2, (long)NB * RCC, RCC, NB, RCC, NB, 0, BB * Tout);
            gemm(stream, t2, 0, RCC, KBIG, 0, 0, KBIG, 0,
                 gcW + (long)l * RCC * GINC + (64 + 64 * s), 0, 1, GINC, nullptr,
                 xn, 0, RCC, Mc, RCC, RCC, 1, 1);
        }

        // residual add (end-aligned) + BN stats, then BN apply -> next x
        k_zero<<<ew(128), 256, 0, stream>>>(stats, 128);
        k_resid_stats<<<gs((long)Mc * RCC), 256, 0, stream>>>(
            xn, xcur, Tout * NB, (Tin - Tout) * NB, (Tin - Tout) * NB, Mc * RCC, stats);
        k_bn_apply<<<ew((long)Mc * RCC), 256, 0, stream>>>(
            xn, stats, bnG + l * 32, bnB + l * 32, (float)Mc, xnxt, Mc * RCC);

        float* tmp = xcur; xcur = xnxt; xnxt = tmp;
        Tin = Tout;
    }

    // --- head ---
    const int MN = BB * NB;
    k_relu<<<ew((long)MN * SCC), 256, 0, stream>>>(skip, t1, MN * SCC);   // xr = relu(skip)
    // q = xr @ Wq  (Wq is [c,d] -> transposed weight access)
    gemm(stream, t1, 0, SCC, KBIG, 0, 0, KBIG, 0,
         Wq, 0, SCC, 1, nullptr, xg, 0, SCC, MN, SCC, SCC, 0, 1);
    k_kmat<<<ew(NPOI * SCC), 256, 0, stream>>>(poiVec, Wk, kmat);
    k_attn<<<ew(MN), 256, 0, stream>>>(xg, kmat, poi,
                                       out + (long)BB * OUTC * NB,
                                       out + (long)BB * OUTC * NB + (long)MN * NPOI, pobuf);
    k_scale<<<ew((long)MN * SCC), 256, 0, stream>>>(t1, pobuf, t2, MN * SCC);
    // end1 (relu) and end2
    gemm(stream, t2, 0, SCC, KBIG, 0, 0, KBIG, 0,
         e1W, 0, 1, SCC, e1B, e1buf, 0, ECC, MN, ECC, SCC, 2, 1);
    gemm(stream, e1buf, 0, ECC, KBIG, 0, 0, KBIG, 0,
         e2W, 0, 1, ECC, e2B, e2buf, 0, OUTC, MN, OUTC, ECC, 0, 1);
    k_outT<<<ew((long)BB * OUTC * NB), 256, 0, stream>>>(e2buf, out, BB * OUTC * NB);
}
